// GeoSample3D_78013785964618
// MI455X (gfx1250) — matrix operand order repaired and checked by reading, not compiled, not run on hardware
//
#include <hip/hip_runtime.h>
#include <hip/hip_bf16.h>
#include <math.h>

// ---------------- problem constants (fixed by reference) ----------------
#define NN   2
#define CC   32
#define DD   48
#define HH   48
#define WW   48
#define KK   3
#define TT   8            // 1 + K + 3 + 1
#define TC   256          // T * C
#define HW   (HH*WW)      // 2304
#define DHW  (DD*HH*WW)   // 110592
#define NVOX (NN*DHW)     // 221184
#define EPSF 1e-6f

#define BLOCK_VOX 64      // voxels per block == threads per block (2 waves)
// DHW % BLOCK_VOX == 0  -> a block never straddles two images; n is uniform.

typedef _Float16 half8  __attribute__((ext_vector_type(8)));
typedef _Float16 half16 __attribute__((ext_vector_type(16)));
typedef float    float8 __attribute__((ext_vector_type(8)));

static __device__ __forceinline__ half16 h16cat(half8 lo, half8 hi) {
    union { half16 v; half8 h[2]; } u;
    u.h[0] = lo; u.h[1] = hi;
    return u.v;
}

static __device__ __forceinline__ float sigmoidf_(float v) {
    return 1.0f / (1.0f + __expf(-v));
}

// reflect per reference: c=|coord+0.5|; m=fmod(c,2s); min(m,2s-m)-0.5
static __device__ __forceinline__ float reflectf_(float coord, float size) {
    float c = fabsf(coord + 0.5f);
    float m = fmodf(c, 2.0f * size);
    return fminf(m, 2.0f * size - m) - 0.5f;
}

struct Samp {
    int o00, o01, o10, o11;   // (z,y) plane offsets: (z0,y0),(z0,y1),(z1,y0),(z1,y1)
    int x0, x1;
    float fx, fy, fz;
};

static __device__ __forceinline__ void make_samp(float gx, float gy, float gz, Samp& s) {
    float ix = reflectf_(((gx + 1.0f) * (float)WW - 1.0f) * 0.5f, (float)WW);
    float iy = reflectf_(((gy + 1.0f) * (float)HH - 1.0f) * 0.5f, (float)HH);
    float iz = reflectf_(((gz + 1.0f) * (float)DD - 1.0f) * 0.5f, (float)DD);
    float x0f = floorf(ix), y0f = floorf(iy), z0f = floorf(iz);
    s.fx = ix - x0f; s.fy = iy - y0f; s.fz = iz - z0f;
    int x0i = min(max((int)x0f, 0), WW - 1); int x1i = min(x0i + 1, WW - 1);
    int y0i = min(max((int)y0f, 0), HH - 1); int y1i = min(y0i + 1, HH - 1);
    int z0i = min(max((int)z0f, 0), DD - 1); int z1i = min(z0i + 1, DD - 1);
    s.o00 = z0i * HW + y0i * WW;
    s.o01 = z0i * HW + y1i * WW;
    s.o10 = z1i * HW + y0i * WW;
    s.o11 = z1i * HW + y1i * WW;
    s.x0 = x0i; s.x1 = x1i;
}

static __device__ __forceinline__ float trilin(const float* __restrict__ base, const Samp& s) {
    float v000 = base[s.o00 + s.x0], v001 = base[s.o00 + s.x1];
    float v010 = base[s.o01 + s.x0], v011 = base[s.o01 + s.x1];
    float v100 = base[s.o10 + s.x0], v101 = base[s.o10 + s.x1];
    float v110 = base[s.o11 + s.x0], v111 = base[s.o11 + s.x1];
    float c00 = v000 * (1.0f - s.fx) + v001 * s.fx;
    float c01 = v010 * (1.0f - s.fx) + v011 * s.fx;
    float c10 = v100 * (1.0f - s.fx) + v101 * s.fx;
    float c11 = v110 * (1.0f - s.fx) + v111 * s.fx;
    float c0  = c00 * (1.0f - s.fy) + c01 * s.fy;
    float c1  = c10 * (1.0f - s.fy) + c11 * s.fy;
    return c0 * (1.0f - s.fz) + c1 * s.fz;
}

__global__ __launch_bounds__(BLOCK_VOX)
void GeoSample3D_fused_kernel(const float* __restrict__ x,
                              const float* __restrict__ w_field,
                              const float* __restrict__ b_field,
                              const float* __restrict__ gates,
                              const float* __restrict__ w_mix,
                              const float* __restrict__ b_mix,
                              float* __restrict__ out) {
    // LDS: gated tokens (f16) for 64 voxels, and w_mix converted to f16
    __shared__ __align__(16) _Float16 tokLds[BLOCK_VOX * TC];  // 32 KB
    __shared__ __align__(16) _Float16 wmixLds[CC * TC];        // 16 KB

    const int tid = threadIdx.x;
    const int blockVoxBase = blockIdx.x * BLOCK_VOX;
    // Uniform (scalar-path) image index + spatial base: DHW % BLOCK_VOX == 0.
    const int n       = blockVoxBase / DHW;    // SALU, uniform
    const int spBase  = blockVoxBase % DHW;    // SALU, uniform
    const float* __restrict__ xn = x + (size_t)n * CC * DHW;

    // ---- stage w_mix (fp32 -> f16) into LDS ----
    for (int i = tid; i < CC * TC; i += BLOCK_VOX)
        wmixLds[i] = (_Float16)w_mix[i];

    // =====================================================================
    // Phase 1: per-thread voxel -> 256 gated tokens (f16) in LDS
    // =====================================================================
    {
        const int sp = spBase + tid;          // < DHW by construction
        const int zc = sp / HW;
        const int rm = sp % HW;
        const int yc = rm / WW;
        const int xc = rm % WW;

        // gate sigmoids
        float gs[TT];
        #pragma unroll
        for (int t = 0; t < TT; ++t) gs[t] = sigmoidf_(gates[t]);

        // load the 32 channels of this voxel
        float xi[CC];
        #pragma unroll
        for (int c = 0; c < CC; ++c) xi[c] = xn[(size_t)c * DHW + sp];

        // field projection: f[12] = w_field(12x32) * xi + b_field
        float f[4 * KK];
        #pragma unroll
        for (int j = 0; j < 4 * KK; ++j) {
            float acc = b_field[j];
            const float* wr = w_field + j * CC;
            #pragma unroll
            for (int c = 0; c < CC; ++c) acc = fmaf(wr[c], xi[c], acc);
            f[j] = acc;
        }

        // per-k direction / radius, and 6 sample descriptors
        float ux[KK], uy[KK], uz[KK], invr[KK];
        Samp sp_p[KK], sp_m[KK];
        const float gx0 = ((float)xc + 0.5f) * 2.0f / (float)WW - 1.0f;
        const float gy0 = ((float)yc + 0.5f) * 2.0f / (float)HH - 1.0f;
        const float gz0 = ((float)zc + 0.5f) * 2.0f / (float)DD - 1.0f;
        #pragma unroll
        for (int k = 0; k < KK; ++k) {
            float vx = f[4 * k + 0], vy = f[4 * k + 1], vz = f[4 * k + 2];
            float invn = 1.0f / sqrtf(vx * vx + vy * vy + vz * vz + EPSF);
            ux[k] = vx * invn; uy[k] = vy * invn; uz[k] = vz * invn;
            float r = 0.5f + 1.5f * sigmoidf_(f[4 * k + 3]);
            invr[k] = 1.0f / (r + EPSF);
            float dx = 2.0f * r * ux[k] / ((float)WW + EPSF);
            float dy = 2.0f * r * uy[k] / ((float)HH + EPSF);
            float dz = 2.0f * r * uz[k] / ((float)DD + EPSF);
            make_samp(gx0 + dx, gy0 + dy, gz0 + dz, sp_p[k]);
            make_samp(gx0 - dx, gy0 - dy, gz0 - dz, sp_m[k]);
        }

        _Float16* __restrict__ trow = tokLds + tid * TC;
        const float third = 1.0f / 3.0f;
        for (int c = 0; c < CC; ++c) {
            const float* __restrict__ base = xn + (size_t)c * DHW;
            float G0 = 0.0f, G1 = 0.0f, G2 = 0.0f, Lc = 0.0f;
            float av[KK];
            #pragma unroll
            for (int k = 0; k < KK; ++k) {
                float xp = trilin(base, sp_p[k]);
                float xm = trilin(base, sp_m[k]);
                float sum = xp + xm;
                av[k] = 0.5f * sum;
                float okc = 0.5f * (xp - xm) * invr[k];
                G0 = fmaf(ux[k], okc, G0);
                G1 = fmaf(uy[k], okc, G1);
                G2 = fmaf(uz[k], okc, G2);
                Lc = fmaf(sum - 2.0f * xi[c], invr[k] * invr[k], Lc);
            }
            Lc *= third;
            // tokens: [x, a0..a2, G0..G2, L], gated
            trow[0 * CC + c] = (_Float16)(xi[c] * gs[0]);
            trow[1 * CC + c] = (_Float16)(av[0] * gs[1]);
            trow[2 * CC + c] = (_Float16)(av[1] * gs[2]);
            trow[3 * CC + c] = (_Float16)(av[2] * gs[3]);
            trow[4 * CC + c] = (_Float16)(G0 * gs[4]);
            trow[5 * CC + c] = (_Float16)(G1 * gs[5]);
            trow[6 * CC + c] = (_Float16)(G2 * gs[6]);
            trow[7 * CC + c] = (_Float16)(Lc * gs[7]);
        }
    }

    __syncthreads();

    // =====================================================================
    // Phase 2: WMMA GEMM  delta[oc, vox] = w_mix(32x256) . tokens(256 x 64)
    //   A (16x32 f16): M = out-channel, K = token-channel
    //   B (32x16 f16): K = token-channel, N = voxel (lane axis -> coalesced)
    // =====================================================================
    {
        const int lane  = tid & 31;
        const int wave  = tid >> 5;
        const int lhalf = lane >> 4;   // 0: lanes 0-15, 1: lanes 16-31
        const int lcol  = lane & 15;

        // per-lane invariant pieces of the store address
        const float* __restrict__ xrow = xn;            // + (oc*DHW + sp)
        float* __restrict__ orow = out + (size_t)n * CC * DHW;

        for (int nt = wave; nt < BLOCK_VOX / 16; nt += 2) {
            float8 acc[2] = { {}, {} };
            #pragma unroll
            for (int ks = 0; ks < TC / 32; ++ks) {
                const int kbase = ks * 32;
                // B fragment: lane holds column N=lcol, K = lhalf*16 .. +15 (contiguous)
                const _Float16* bp = tokLds + (nt * 16 + lcol) * TC + kbase + lhalf * 16;
                half16 B = h16cat(*(const half8*)bp, *(const half8*)(bp + 8));
                #pragma unroll
                for (int mt = 0; mt < 2; ++mt) {
                    // A fragment: lane holds row M=lcol, K = lhalf*16 .. +15 (contiguous)
                    const _Float16* ap = wmixLds + (mt * 16 + lcol) * TC + kbase + lhalf * 16;
                    half16 A = h16cat(*(const half8*)ap, *(const half8*)(ap + 8));
                    acc[mt] = __builtin_amdgcn_wmma_f32_16x16x32_f16(
                        false, A, false, B, (short)0, acc[mt], false, false);
                }
            }
            // epilogue: out = x + b_mix + delta   (lanes -> consecutive voxels;
            // n/spBase are uniform, so addressing is lane-add + immediate only)
            const int sp = spBase + nt * 16 + lcol;
            #pragma unroll
            for (int mt = 0; mt < 2; ++mt) {
                const int ocb = mt * 16 + lhalf * 8;
                const size_t base = (size_t)ocb * DHW + sp;
                #pragma unroll
                for (int r = 0; r < 8; ++r) {
                    const size_t idx = base + (size_t)r * DHW;
                    orow[idx] = xrow[idx] + b_mix[ocb + r] + acc[mt][r];
                }
            }
        }
    }
}

extern "C" void kernel_launch(void* const* d_in, const int* in_sizes, int n_in,
                              void* d_out, int out_size, void* d_ws, size_t ws_size,
                              hipStream_t stream) {
    (void)in_sizes; (void)n_in; (void)out_size; (void)d_ws; (void)ws_size;
    const float* x       = (const float*)d_in[0];
    const float* w_field = (const float*)d_in[1];
    const float* b_field = (const float*)d_in[2];
    const float* gates   = (const float*)d_in[3];
    const float* w_mix   = (const float*)d_in[4];
    const float* b_mix   = (const float*)d_in[5];
    float* out = (float*)d_out;

    dim3 grid(NVOX / BLOCK_VOX);   // 3456 blocks
    dim3 block(BLOCK_VOX);         // 64 threads = 2 waves (wave32)
    GeoSample3D_fused_kernel<<<grid, block, 0, stream>>>(
        x, w_field, b_field, gates, w_mix, b_mix, out);
}
